// AttentionHead_1005022347433
// MI455X (gfx1250) — matrix operand order repaired and checked by reading, not compiled, not run on hardware
//
#include <hip/hip_runtime.h>
#include <stdint.h>

typedef _Float16 v16h __attribute__((ext_vector_type(16)));
typedef _Float16 v8h  __attribute__((ext_vector_type(8)));
typedef float    v8f  __attribute__((ext_vector_type(8)));
typedef unsigned int u32x4 __attribute__((ext_vector_type(4)));
typedef int          i32x8 __attribute__((ext_vector_type(8)));
typedef int          i32x4 __attribute__((ext_vector_type(4)));

#define TQ 4096
#define CH 512
#define HD 64
#define NB 4

static __device__ __forceinline__ v8f wmma_f16(v16h a, v16h b, v8f c) {
  return __builtin_amdgcn_wmma_f32_16x16x32_f16(false, a, false, b,
                                                (short)0, c, false, false);
}

static __device__ __forceinline__ uint32_t lds_off(const void* p) {
  // generic LDS address: low 32 bits are the LDS byte offset
  return (uint32_t)(uintptr_t)p;
}

// CDNA5 async copy: global -> LDS, 16 bytes per lane, tracked by ASYNCcnt
static __device__ __forceinline__ void async_cp16(void* lds_dst, const void* gsrc) {
  asm volatile("global_load_async_to_lds_b128 %0, %1, off"
               :: "v"(lds_off(lds_dst)), "v"(gsrc) : "memory");
}
#define WAIT_ASYNC(n) asm volatile("s_wait_asynccnt " #n ::: "memory")
static __device__ __forceinline__ void wait_ds0() {
  asm volatile("s_wait_dscnt 0" ::: "memory");
}

// CDNA5 Tensor Data Mover: 2D tile DMA global -> LDS (D# per ISA 08_async_tensor)
// clang-23 toolchain: 6-arg builtin (g0, g1, g2, g3, g4, cpol)
static __device__ __forceinline__ void tdm_load_2d(
    void* lds_dst, const void* gsrc,
    uint32_t tensor_d0, uint32_t tensor_d1,
    uint32_t tile_d0, uint32_t tile_d1, uint32_t d0_stride)
{
  const uint64_t ga = (uint64_t)(uintptr_t)gsrc;
  u32x4 g0 = { 1u,                                   // count=1, user descriptor
               lds_off(lds_dst),                     // lds_addr [63:32]
               (uint32_t)ga,                         // global_addr [95:64]
               (uint32_t)((ga >> 32) & 0x01FFFFFFu)  // global_addr [120:96]
                 | 0x80000000u };                    // type=2 ("image") [127:126]
  i32x8 g1;
  g1[0] = 0x00010000;                                   // data_size=1 -> 2 bytes
  g1[1] = (int)(tensor_d0 << 16);                       // tensor_dim0 lo16
  g1[2] = (int)((tensor_d0 >> 16) | (tensor_d1 << 16)); // dim0 hi | dim1 lo
  g1[3] = (int)((tensor_d1 >> 16) | (tile_d0 << 16));   // dim1 hi | tile_dim0
  g1[4] = (int)tile_d1;                                 // tile_dim1 | tile_dim2=0
  g1[5] = (int)d0_stride;                               // tensor_dim0_stride lo32
  g1[6] = 0;                                            // stride hi | dim1_stride lo
  g1[7] = 0;
  i32x4 z4 = { 0, 0, 0, 0 };
  i32x8 z8 = { 0, 0, 0, 0, 0, 0, 0, 0 };
  __builtin_amdgcn_tensor_load_to_lds(g0, g1, z4, z4, z8, 0);
}

static __device__ __forceinline__ v16h ld16h(const _Float16* p) {
  v8h lo = *(const v8h*)p;
  v8h hi = *(const v8h*)(p + 8);
  v16h r;
#pragma unroll
  for (int j = 0; j < 8; ++j) { r[j] = lo[j]; r[j + 8] = hi[j]; }
  return r;
}

// ---------------- Kernel 1: projections x@W -> f16 q, k, v^T ----------------
__global__ __launch_bounds__(128) void proj_kernel(
    const float* __restrict__ x,
    const float* __restrict__ Wq,
    const float* __restrict__ Wk,
    const float* __restrict__ Wv,
    _Float16* __restrict__ qh,      // [B*T, H]
    _Float16* __restrict__ kh,      // [B*T, H]
    _Float16* __restrict__ vt)      // [B, H, T] (transposed)
{
  const int tile = blockIdx.x;
  const int mat  = blockIdx.y;              // 0:q 1:k 2:v
  const int wave = threadIdx.x >> 5;        // n-chunk 0..3
  const int lane = threadIdx.x & 31;
  const int half = lane >> 4;
  const int l16  = lane & 15;

  const float* W = (mat == 0) ? Wq : (mat == 1) ? Wk : Wv;
  const int row0 = tile * 16;
  const int nb   = wave * 16;

  const float* xrow = x + (size_t)(row0 + l16) * CH;
  v8f acc = {};

  for (int kc = 0; kc < CH; kc += 32) {
    // A tile (x) in A-layout: row = l16, K(j) = 16*(j>>3) + 8*half + (j&7)
    v16h a;
    {
      const float* pa0 = xrow + kc + half * 8;
      const float* pa1 = xrow + kc + 16 + half * 8;
#pragma unroll
      for (int j = 0; j < 8; ++j) {
        a[j]     = (_Float16)pa0[j];
        a[j + 8] = (_Float16)pa1[j];
      }
    }
    // B tile (W) in B-layout: col = nb+l16, K(j) = 16*(j>>3) + 8*half + (j&7)
    v16h bm;
    {
      const float* pb = W + (size_t)(kc + half * 8) * HD + nb + l16;
#pragma unroll
      for (int j = 0; j < 16; ++j) bm[j] = (_Float16)pb[(size_t)(j + 8 * (j >> 3)) * HD];
    }
    acc = wmma_f16(a, bm, acc);
  }

  // D-layout: M = 8*half + r, N = nb + l16   (uniform branch hoisted out)
  if (mat != 2) {
    _Float16* dst = (mat == 0) ? qh : kh;
#pragma unroll
    for (int r = 0; r < 8; ++r) {
      const int rowg = row0 + half * 8 + r;
      dst[(size_t)rowg * HD + nb + l16] = (_Float16)acc[r];
    }
  } else {
    const int bidx = row0 >> 12;            // / 4096 (tile never crosses batch)
    const int t0   = row0 & (TQ - 1);
    _Float16* col = vt + ((size_t)(bidx * HD + nb + l16)) * TQ;
#pragma unroll
    for (int r = 0; r < 8; ++r)
      col[t0 + half * 8 + r] = (_Float16)acc[r];
  }
}

// ---------------- Kernel 2: causal flash attention ----------------
// grid (T/64, B), block 128 (4 waves x 16 queries). Double-buffered staging:
// K tiles via async-LDS copies, V^T tiles via TDM (one descriptor per tile).
__global__ __launch_bounds__(128) void attn_kernel(
    const _Float16* __restrict__ qh,
    const _Float16* __restrict__ kh,
    const _Float16* __restrict__ vt,
    float* __restrict__ out)
{
  __shared__ _Float16 Kt[2][32 * 64];     // [key][h]
  __shared__ _Float16 Vs[2][64 * 32];     // [h][key]   (V^T tile)
  __shared__ _Float16 Pt[4][16 * 32];     // per-wave P bounce buffer

  const int b    = blockIdx.y;
  const int q0   = blockIdx.x * 64;
  const int tid  = threadIdx.x;
  const int wave = tid >> 5;
  const int lane = tid & 31;
  const int half = lane >> 4;
  const int l16  = lane & 15;
  const int qw   = q0 + wave * 16;

  const _Float16* qg = qh + (size_t)(b * TQ) * HD;
  const _Float16* kg = kh + (size_t)(b * TQ) * HD;
  const _Float16* vg = vt + (size_t)(b * HD) * TQ;

  // Q tile in A-layout for both 32-wide h-chunks (loaded once)
  v16h qa[2];
  {
    const _Float16* qrow = qg + (size_t)(qw + l16) * HD;
#pragma unroll
    for (int c = 0; c < 2; ++c) {
      v8h lo = *(const v8h*)(qrow + c * 32 + half * 16);
      v8h hi = *(const v8h*)(qrow + c * 32 + half * 16 + 8);
#pragma unroll
      for (int j = 0; j < 8; ++j) { qa[c][j] = lo[j]; qa[c][j + 8] = hi[j]; }
    }
  }

  // staging helpers
  auto stageK = [&](int buf, int kb) {
#pragma unroll
    for (int i = 0; i < 2; ++i) {
      const int seg  = tid + i * 128;       // 256 x 16B segments
      const int krow = seg >> 3;
      const int koff = (seg & 7) * 8;
      async_cp16(&Kt[buf][krow * 64 + koff],
                 kg + (size_t)(kb + krow) * HD + koff);
    }
  };
  auto stageV = [&](int buf, int kb) {
    if (wave == 0)  // one TDM descriptor moves the whole 32x64 tile
      tdm_load_2d(&Vs[buf][0], vg + kb, TQ, HD, /*tile*/32, HD, /*stride*/TQ);
  };

  v8f o0 = {}, o1 = {}, o2 = {}, o3 = {};
  float mrow[8], lrow[8];
#pragma unroll
  for (int r = 0; r < 8; ++r) { mrow[r] = -__builtin_inff(); lrow[r] = 0.0f; }

  const float scale = 0.125f;               // H^-0.5, H=64
  const int nkb = (q0 + 64) >> 5;           // causal block limit

  stageK(0, 0);
  stageV(0, 0);

  for (int kbi = 0; kbi < nkb; ++kbi) {
    const int kb  = kbi << 5;
    const int cur = kbi & 1;

    __syncthreads();                        // all waves done with buf[1-cur]
    if (kbi + 1 < nkb) {                    // prefetch next block
      stageK(1 - cur, kb + 32);
      stageV(1 - cur, kb + 32);
      WAIT_ASYNC(2);                        // cur K done (in-order), next in flight
      if (wave == 0) __builtin_amdgcn_s_wait_tensorcnt((short)1);
    } else {
      WAIT_ASYNC(0);
      if (wave == 0) __builtin_amdgcn_s_wait_tensorcnt((short)0);
    }
    __syncthreads();                        // cur buffers visible to all waves

    if (kb <= qw + 15) {                    // wave-uniform causal skip
      // S = Q K^T : keys 0..15 (slo) and 16..31 (shi) of this block
      v8f slo = {}, shi = {};
#pragma unroll
      for (int c = 0; c < 2; ++c) {
        v16h bl = ld16h(&Kt[cur][l16 * 64 + c * 32 + half * 16]);
        v16h bh = ld16h(&Kt[cur][(16 + l16) * 64 + c * 32 + half * 16]);
        slo = wmma_f16(qa[c], bl, slo);
        shi = wmma_f16(qa[c], bh, shi);
      }

      // online softmax (rows M = 8*half + r ; N across 16-lane halves)
      float plo[8], phi[8], alpha[8];
#pragma unroll
      for (int r = 0; r < 8; ++r) {
        const int qidx = qw + half * 8 + r;
        float sl = slo[r] * scale;
        float sh = shi[r] * scale;
        if (kb + l16 > qidx)      sl = -__builtin_inff();
        if (kb + 16 + l16 > qidx) sh = -__builtin_inff();
        float mx = fmaxf(sl, sh);
#pragma unroll
        for (int d = 8; d >= 1; d >>= 1) mx = fmaxf(mx, __shfl_xor(mx, d, 32));
        const float mnew = fmaxf(mrow[r], mx);
        const float a  = __expf(mrow[r] - mnew);
        const float pl = __expf(sl - mnew);
        const float ph = __expf(sh - mnew);
        float rs = pl + ph;
#pragma unroll
        for (int d = 8; d >= 1; d >>= 1) rs += __shfl_xor(rs, d, 32);
        lrow[r] = lrow[r] * a + rs;
        mrow[r] = mnew;
        alpha[r] = a;
        plo[r] = pl; phi[r] = ph;
      }
#pragma unroll
      for (int r = 0; r < 8; ++r) {
        o0[r] *= alpha[r]; o1[r] *= alpha[r];
        o2[r] *= alpha[r]; o3[r] *= alpha[r];
      }

      // D-layout -> A-layout transpose of P through per-wave LDS
      _Float16* pw = &Pt[wave][0];
#pragma unroll
      for (int r = 0; r < 8; ++r) {
        const int m = half * 8 + r;
        pw[m * 32 + l16]      = (_Float16)plo[r];
        pw[m * 32 + 16 + l16] = (_Float16)phi[r];
      }
      wait_ds0();
      v16h pa;
      {
        const _Float16* pr = pw + l16 * 32;
        v8h lo = *(const v8h*)(pr + half * 16);
        v8h hi = *(const v8h*)(pr + half * 16 + 8);
#pragma unroll
        for (int j = 0; j < 8; ++j) { pa[j] = lo[j]; pa[j + 8] = hi[j]; }
      }

      // O += P @ V : B-operand rows contiguous in the V^T tile
      v16h bv;
      bv = ld16h(&Vs[cur][( 0 + l16) * 32 + half * 16]); o0 = wmma_f16(pa, bv, o0);
      bv = ld16h(&Vs[cur][(16 + l16) * 32 + half * 16]); o1 = wmma_f16(pa, bv, o1);
      bv = ld16h(&Vs[cur][(32 + l16) * 32 + half * 16]); o2 = wmma_f16(pa, bv, o2);
      bv = ld16h(&Vs[cur][(48 + l16) * 32 + half * 16]); o3 = wmma_f16(pa, bv, o3);
    }
  }

  // epilogue: out[b][qw+m][h] = O / l
  float* og = out + ((size_t)(b * TQ + qw)) * HD;
#pragma unroll
  for (int r = 0; r < 8; ++r) {
    const int m = half * 8 + r;
    const float inv = 1.0f / lrow[r];
    og[(size_t)m * HD +  0 + l16] = o0[r] * inv;
    og[(size_t)m * HD + 16 + l16] = o1[r] * inv;
    og[(size_t)m * HD + 32 + l16] = o2[r] * inv;
    og[(size_t)m * HD + 48 + l16] = o3[r] * inv;
  }
}

extern "C" void kernel_launch(void* const* d_in, const int* in_sizes, int n_in,
                              void* d_out, int out_size, void* d_ws, size_t ws_size,
                              hipStream_t stream) {
  const float* x  = (const float*)d_in[0];
  const float* Wk = (const float*)d_in[1];
  const float* Wq = (const float*)d_in[2];
  const float* Wv = (const float*)d_in[3];
  float* out = (float*)d_out;

  char* ws = (char*)d_ws;
  const size_t qk_bytes = (size_t)NB * TQ * HD * sizeof(_Float16);  // 2 MB each
  _Float16* qh = (_Float16*)ws;
  _Float16* kh = (_Float16*)(ws + qk_bytes);
  _Float16* vt = (_Float16*)(ws + 2 * qk_bytes);

  proj_kernel<<<dim3(NB * TQ / 16, 3), 128, 0, stream>>>(x, Wq, Wk, Wv, qh, kh, vt);
  attn_kernel<<<dim3(TQ / 64, NB), 128, 0, stream>>>(qh, kh, vt, out);
}
